// DecoderBlock_54030688583983
// MI455X (gfx1250) — compile-verified, matrix operand order checked by reading, not run on hardware
//
#include <hip/hip_runtime.h>

typedef __bf16 bf16;
typedef __attribute__((ext_vector_type(16))) __bf16   v16bf;
typedef __attribute__((ext_vector_type(8)))  float    v8f;
typedef __attribute__((ext_vector_type(8)))  unsigned v8u;

#define DEV static __device__ __forceinline__

DEV v8f wmma_bf16f32(v16bf a, v16bf b, v8f c) {
  // D = A(16x32 bf16) * B(32x16 bf16) + C(16x16 f32)
  return __builtin_amdgcn_wmma_f32_16x16x32_bf16(false, a, false, b, (short)0, c, false, false);
}

// 16x32 bf16 fragment from row-major source (leading dim `ld`, elems).
// lane&15 = row; VGPR j holds the ISA K-pair with the lane-half +8 swizzle.
// Pairs are contiguous -> merges into ds_load_b128 when ld*2 % 16 == 0.
DEV v16bf load_frag_row(const bf16* base, int ld) {
  const int lane = threadIdx.x & 31;
  const int m = lane & 15, hh = lane >> 4;
  v8u w;
#pragma unroll
  for (int j = 0; j < 8; ++j) {
    const int kb = ((j < 4) ? (2 * j) : (16 + 2 * (j - 4))) + 8 * hh;
    w[j] = *(const unsigned*)(base + (long)m * ld + kb);
  }
  return __builtin_bit_cast(v16bf, w);
}

DEV v16bf ones_frag() {  // bf16 1.0 = 0x3F80, packed pairs
  v8u w;
#pragma unroll
  for (int j = 0; j < 8; ++j) w[j] = 0x3F803F80u;
  return __builtin_bit_cast(v16bf, w);
}

// CDNA5 async global->LDS copy (ASYNCcnt-tracked), 16B per lane.
DEV void async_copy_b128(unsigned lds_off, unsigned long long ga) {
  asm volatile("global_load_async_to_lds_b128 %0, %1, off"
               :: "v"(lds_off), "v"(ga) : "memory");
}
DEV void async_wait0() { asm volatile("s_wait_asynccnt 0x0" ::: "memory"); }
DEV unsigned lds_lo32(const void* p) { return (unsigned)(size_t)p; }  // LDS aperture: addr[31:0]

// ---------------------------------------------------------------------------
// bf16 GEMM: C[M,N] = act(A[M,K](bf16) * W(f32->bf16) + bias[N])
// 256 thr = 8 waves (2x4), tile 128x128, K-step 64, DOUBLE-BUFFERED LDS:
// after the top-of-stage barrier we issue the next A tile via
// global_load_async_to_lds_b128 and the next W tile fill (float4 loads,
// f32->bf16, transposed [n][k] ld=72) into the other buffer, then run 16
// WMMAs on the current buffer. One barrier + one s_wait_asynccnt per stage.
// WMODE: 0 = W row-major [K,N] (ldw=N); 1 = per-head interleave W[H][K][64]
// OUT_MODE: 0 = f32 row-major, 1 = bf16 row-major, 2 = bf16 [B,H,S,64]
// ACT: 0 = none, 1 = relu
// ---------------------------------------------------------------------------
template <int WMODE, int OUT_MODE, int ACT>
__global__ __launch_bounds__(256) void gemm_kernel(
    const bf16* __restrict__ A, int lda,
    const float* __restrict__ W, int ldw, long whs,
    const float* __restrict__ bias,
    void* __restrict__ Cout, int M, int N, int K) {
  __shared__ __align__(16) bf16 ldsA[2][128 * 64];   // [m][k]
  __shared__ __align__(16) bf16 ldsBt[2][128 * 72];  // [n][k] padded
  const int tid = threadIdx.x;
  const int wave = tid >> 5;
  const int wm = wave >> 2;  // 0..1 -> 64 rows each
  const int wn = wave & 3;   // 0..3 -> 32 cols each
  const int m0 = blockIdx.y * 128;
  const int n0 = blockIdx.x * 128;

  v8f zero = {};
  v8f acc[4][2];
#pragma unroll
  for (int i = 0; i < 4; ++i) { acc[i][0] = zero; acc[i][1] = zero; }

  const unsigned long long Ag = (unsigned long long)(size_t)A;

  auto stage = [&](int ks, int pp) {
    // A tile 128x64 bf16 = 16 KB: 4 async b128 chunks/thread.
    const unsigned labase = lds_lo32(ldsA[pp]);
#pragma unroll
    for (int i = 0; i < 4; ++i) {
      int cidx = tid + 256 * i;
      int r = cidx >> 3, ck = cidx & 7;  // 8 chunks of 16B per 128B row
      unsigned long long ga = Ag + ((long)(m0 + r) * lda + ks) * 2 + 16 * ck;
      async_copy_b128(labase + 16u * cidx, ga);
    }
    // W tile 64x128 f32 -> bf16, transposed into ldsBt[pp][n][k]; float4 loads.
    bf16* bt = ldsBt[pp];
#pragma unroll
    for (int it = 0; it < 8; ++it) {
      int qidx = tid + 256 * it;               // 2048 quads
      int r = qidx >> 5, c4 = (qidx & 31) * 4; // r = k, c4 = n quad
      float4 wv;
      if (WMODE == 0) {
        wv = *(const float4*)(W + (long)(ks + r) * ldw + n0 + c4);
      } else {
        int cc = n0 + c4;  // quads never straddle a 64-col head boundary
        wv = *(const float4*)(W + (long)(cc >> 6) * whs + (long)(ks + r) * 64 + (cc & 63));
      }
      bt[(long)(c4 + 0) * 72 + r] = (bf16)wv.x;
      bt[(long)(c4 + 1) * 72 + r] = (bf16)wv.y;
      bt[(long)(c4 + 2) * 72 + r] = (bf16)wv.z;
      bt[(long)(c4 + 3) * 72 + r] = (bf16)wv.w;
    }
  };

  stage(0, 0);
  int p = 0;
  for (int k0 = 0; k0 < K; k0 += 64) {
    async_wait0();    // my async chunks for buffer p have landed
    __syncthreads();  // everyone's fills of p done; everyone's reads of p^1 done
    if (k0 + 64 < K) stage(k0 + 64, p ^ 1);
    // 12 fragment loads up front, then 16 WMMAs
    v16bf af[2][4], bfr[2][2];
#pragma unroll
    for (int s = 0; s < 2; ++s) {
#pragma unroll
      for (int i = 0; i < 4; ++i)
        af[s][i] = load_frag_row(ldsA[p] + (long)(wm * 64 + 16 * i) * 64 + s * 32, 64);
#pragma unroll
      for (int j = 0; j < 2; ++j)
        bfr[s][j] = load_frag_row(ldsBt[p] + (long)(wn * 32 + 16 * j) * 72 + s * 32, 72);
    }
#pragma unroll
    for (int s = 0; s < 2; ++s)
#pragma unroll
      for (int i = 0; i < 4; ++i)
#pragma unroll
        for (int j = 0; j < 2; ++j)
          acc[i][j] = wmma_bf16f32(af[s][i], bfr[s][j], acc[i][j]);
    p ^= 1;
  }

  const int lane = tid & 31;
  const int nI = lane & 15, hh = lane >> 4;
#pragma unroll
  for (int i = 0; i < 4; ++i)
#pragma unroll
    for (int j = 0; j < 2; ++j)
#pragma unroll
      for (int r = 0; r < 8; ++r) {
        int row = m0 + wm * 64 + i * 16 + r + 8 * hh;
        int col = n0 + wn * 32 + j * 16 + nI;
        float vv = acc[i][j][r] + bias[col];
        if (ACT == 1) vv = fmaxf(vv, 0.0f);
        if (OUT_MODE == 0) {
          ((float*)Cout)[(long)row * N + col] = vv;
        } else if (OUT_MODE == 1) {
          ((bf16*)Cout)[(long)row * N + col] = (bf16)vv;
        } else {  // [B,H,S,64]; S=2048, H=8; head from column
          int bb = row >> 11, ss = row & 2047;
          ((bf16*)Cout)[((((long)bb * 8 + (col >> 6)) * 2048) + ss) * 64 + (col & 63)] = (bf16)vv;
        }
      }
}

// ---------------------------------------------------------------------------
// Flash attention, double-buffered K/V: WG = 4 waves, 16 query rows each,
// 32-key blocks. K tile: async b128 to LDS. V tile: transposed [vdim][key]
// (ld=40, uint2 loads). Row sums via an extra WMMA against all-ones; row max
// via a batched (step-outer) ds_bpermute butterfly. 2 barriers per block.
// Post-softmax query mask folded into the final per-row scale.
// Q,K,V: [B,H,S,64] bf16.  O: [B,S,H*64] bf16.
// ---------------------------------------------------------------------------
__global__ __launch_bounds__(128) void attn_kernel(
    const bf16* __restrict__ Q, const bf16* __restrict__ Kt,
    const bf16* __restrict__ V, const float* __restrict__ mask,
    bf16* __restrict__ O) {
  __shared__ __align__(16) bf16 kb[2][32 * 64];  // [key][dk]
  __shared__ __align__(16) bf16 vt[2][64 * 40];  // [vdim][key] padded
  __shared__ __align__(16) bf16 pb[4 * 16 * 32];
  const int tid = threadIdx.x, wave = tid >> 5, lane = tid & 31;
  const int nI = lane & 15, hh = lane >> 4;
  const int bh = blockIdx.z;
  const int b = bh >> 3, h = bh & 7;
  const int q0 = blockIdx.y * 64 + wave * 16;

  const bf16* qbase = Q + ((long)bh * 2048 + q0) * 64;
  v16bf aq0 = load_frag_row(qbase, 64);       // dk 0..31
  v16bf aq1 = load_frag_row(qbase + 32, 64);  // dk 32..63
  const bf16* kg = Kt + (long)bh * 2048 * 64;
  const bf16* vg = V + (long)bh * 2048 * 64;
  const v16bf ones = ones_frag();

  v8f zero = {};
  v8f oacc[4];
  oacc[0] = zero; oacc[1] = zero; oacc[2] = zero; oacc[3] = zero;
  v8f lsum = zero;
  float mrow[8];
#pragma unroll
  for (int r = 0; r < 8; ++r) mrow[r] = -3.0e38f;

  auto stageKV = [&](int js, int pp) {
    unsigned la = lds_lo32(kb[pp]) + 16u * tid;
    unsigned long long gk = (unsigned long long)(size_t)(kg + (long)js * 64);
    async_copy_b128(la, gk + 16u * tid);
    async_copy_b128(la + 2048u, gk + 16u * tid + 2048u);
    const bf16* vrow = vg + (long)js * 64;
    bf16* vtp = vt[pp];
#pragma unroll
    for (int it = 0; it < 4; ++it) {  // transpose V 32x64 -> vt[64][40]
      int qidx = tid + 128 * it;      // 512 quads of 4 bf16
      int r = qidx >> 4, c4 = (qidx & 15) * 4;
      union { uint2 u; unsigned short s[4]; } cv;
      cv.u = *(const uint2*)(vrow + (long)r * 64 + c4);
#pragma unroll
      for (int q = 0; q < 4; ++q)
        *(unsigned short*)&vtp[(long)(c4 + q) * 40 + r] = cv.s[q];
    }
  };

  stageKV(0, 0);
  int p = 0;
  for (int j0 = 0; j0 < 2048; j0 += 32) {
    async_wait0();
    __syncthreads();  // buffer p ready; prior reads of p^1 and of pb done
    if (j0 + 32 < 2048) stageKV(j0 + 32, p ^ 1);

    v8f s0 = zero, s1 = zero;
    {
      const bf16* kp = kb[p];
      v16bf bk0 = load_frag_row(kp, 64);
      v16bf bk1 = load_frag_row(kp + 32, 64);
      v16bf bk2 = load_frag_row(kp + 16 * 64, 64);
      v16bf bk3 = load_frag_row(kp + 16 * 64 + 32, 64);
      s0 = wmma_bf16f32(aq0, bk0, s0);
      s0 = wmma_bf16f32(aq1, bk1, s0);
      s1 = wmma_bf16f32(aq0, bk2, s1);
      s1 = wmma_bf16f32(aq1, bk3, s1);
    }

    // batched row-max butterfly: step-outer so 8 bpermutes issue together
    float v0a[8], v1a[8], mx[8];
#pragma unroll
    for (int r = 0; r < 8; ++r) {
      v0a[r] = s0[r] * 0.125f;  // 1/sqrt(64)
      v1a[r] = s1[r] * 0.125f;
      mx[r] = fmaxf(v0a[r], v1a[r]);
    }
#pragma unroll
    for (int d = 1; d <= 8; d <<= 1) {
#pragma unroll
      for (int r = 0; r < 8; ++r) mx[r] = fmaxf(mx[r], __shfl_xor(mx[r], d, 32));
    }

    bf16* pw = pb + wave * (16 * 32);
#pragma unroll
    for (int r = 0; r < 8; ++r) {
      float nm = fmaxf(mrow[r], mx[r]);
      float sc = __expf(mrow[r] - nm);
      mrow[r] = nm;
      float p0 = __expf(v0a[r] - nm), p1 = __expf(v1a[r] - nm);
      oacc[0][r] = oacc[0][r] * sc; oacc[1][r] = oacc[1][r] * sc;
      oacc[2][r] = oacc[2][r] * sc; oacc[3][r] = oacc[3][r] * sc;
      lsum[r] = lsum[r] * sc;
      pw[(r + 8 * hh) * 32 + nI]      = (bf16)p0;
      pw[(r + 8 * hh) * 32 + 16 + nI] = (bf16)p1;
    }
    __syncthreads();  // publish P (C-layout -> A-layout via LDS)

    v16bf pa = load_frag_row(pw, 32);  // P as 16x32 A fragment
    const bf16* vtp = vt[p];
#pragma unroll
    for (int seg = 0; seg < 4; ++seg) {
      v16bf bv = load_frag_row(vtp + (long)(seg * 16) * 40, 40);
      oacc[seg] = wmma_bf16f32(pa, bv, oacc[seg]);
    }
    lsum = wmma_bf16f32(pa, ones, lsum);  // row sums via P * ones
    p ^= 1;
  }

#pragma unroll
  for (int r = 0; r < 8; ++r) {
    float inv = 1.0f / lsum[r];
    int row = q0 + r + 8 * hh;
    if (mask) inv *= mask[row];  // reference: post-softmax query-row mask
#pragma unroll
    for (int seg = 0; seg < 4; ++seg) {
      int col = h * 64 + seg * 16 + nI;
      O[((long)b * 2048 + row) * 512 + col] = (bf16)(oacc[seg][r] * inv);
    }
  }
}

// ---------------------------------------------------------------------------
// y = LayerNorm(a + res) * g + b ; writes f32 (residual chain) + optional bf16
// ---------------------------------------------------------------------------
__global__ __launch_bounds__(256) void ln_kernel(
    const float* __restrict__ a, const float* __restrict__ res,
    const float* __restrict__ g, const float* __restrict__ be,
    float* __restrict__ yf, bf16* __restrict__ yb) {
  __shared__ float red[256];
  const long row = blockIdx.x;
  const int t = threadIdx.x;
  const float* ar = a + row * 512;
  const float* rr = res + row * 512;
  float x0 = ar[t] + rr[t];
  float x1 = ar[t + 256] + rr[t + 256];
  red[t] = x0 + x1;
  __syncthreads();
  for (int d = 128; d > 0; d >>= 1) { if (t < d) red[t] += red[t + d]; __syncthreads(); }
  float mean = red[0] * (1.0f / 512.0f);
  __syncthreads();
  float d0 = x0 - mean, d1 = x1 - mean;
  red[t] = d0 * d0 + d1 * d1;
  __syncthreads();
  for (int d = 128; d > 0; d >>= 1) { if (t < d) red[t] += red[t + d]; __syncthreads(); }
  float rstd = rsqrtf(red[0] * (1.0f / 512.0f) + 1e-6f);
  float y0 = d0 * rstd * g[t] + be[t];
  float y1 = d1 * rstd * g[t + 256] + be[t + 256];
  yf[row * 512 + t] = y0;
  yf[row * 512 + t + 256] = y1;
  if (yb) { yb[row * 512 + t] = (bf16)y0; yb[row * 512 + t + 256] = (bf16)y1; }
}

__global__ void cvt_kernel(const float* __restrict__ in, bf16* __restrict__ out, long n) {
  long i = (long)blockIdx.x * blockDim.x + threadIdx.x;
  if (i < n) out[i] = (bf16)in[i];
}

// ---------------------------------------------------------------------------
extern "C" void kernel_launch(void* const* d_in, const int* in_sizes, int n_in,
                              void* d_out, int out_size, void* d_ws, size_t ws_size,
                              hipStream_t stream) {
  const float* x    = (const float*)d_in[0];
  const float* enc  = (const float*)d_in[1];
  const float* mask = (const float*)d_in[2];
  const float* Wq_s = (const float*)d_in[3];  const float* bq_s = (const float*)d_in[4];
  const float* Wk_s = (const float*)d_in[5];  const float* bk_s = (const float*)d_in[6];
  const float* Wv_s = (const float*)d_in[7];  const float* bv_s = (const float*)d_in[8];
  const float* Wo_s = (const float*)d_in[9];  const float* bo_s = (const float*)d_in[10];
  const float* Wq_c = (const float*)d_in[11]; const float* bq_c = (const float*)d_in[12];
  const float* Wk_c = (const float*)d_in[13]; const float* bk_c = (const float*)d_in[14];
  const float* Wv_c = (const float*)d_in[15]; const float* bv_c = (const float*)d_in[16];
  const float* Wo_c = (const float*)d_in[17]; const float* bo_c = (const float*)d_in[18];
  const float* ln1g = (const float*)d_in[19]; const float* ln1b = (const float*)d_in[20];
  const float* ln2g = (const float*)d_in[21]; const float* ln2b = (const float*)d_in[22];
  const float* ln3g = (const float*)d_in[23]; const float* ln3b = (const float*)d_in[24];
  const float* W1   = (const float*)d_in[25]; const float* b1   = (const float*)d_in[26];
  const float* W2   = (const float*)d_in[27]; const float* b2   = (const float*)d_in[28];

  // workspace layout (~288 MB)
  char* ws = (char*)d_ws;
  const size_t MB = 1024 * 1024;
  bf16*  xb   = (bf16*)(ws + 0 * MB);    // bf16(x)                 16 MB
  bf16*  eb   = (bf16*)(ws + 16 * MB);   // bf16(encoding)          16 MB
  bf16*  qb   = (bf16*)(ws + 32 * MB);   // Q [B,H,S,64]            16 MB
  bf16*  kbuf = (bf16*)(ws + 48 * MB);   // K                       16 MB
  bf16*  vbuf = (bf16*)(ws + 64 * MB);   // V                       16 MB
  bf16*  ob   = (bf16*)(ws + 80 * MB);   // attn out [B,S,512]      16 MB
  bf16*  hb   = (bf16*)(ws + 96 * MB);   // bf16(h)                 16 MB
  bf16*  ybuf = (bf16*)(ws + 112 * MB);  // bf16(y)                 16 MB
  bf16*  fb   = (bf16*)(ws + 128 * MB);  // relu FFN mid, bf16      64 MB
  float* t0   = (float*)(ws + 192 * MB); // f32 GEMM output         32 MB
  float* hf   = (float*)(ws + 224 * MB); // h (f32)                 32 MB
  float* yf   = (float*)(ws + 256 * MB); // y (f32)                 32 MB

  const long nTok = 16384L * 512;
  dim3 blk128(128), blk256(256);
  dim3 gQKV(4, 128, 1);   // fused heads: N=512
  dim3 gProj(4, 128, 1);  // N=512
  dim3 gFF1(16, 128, 1);  // N=2048
  dim3 gAttn(1, 32, 64);  // S/64=32 query tiles, B*H=64

  // 0. bf16 copies of activations
  cvt_kernel<<<dim3((unsigned)((nTok + 255) / 256)), blk256, 0, stream>>>(x, xb, nTok);
  cvt_kernel<<<dim3((unsigned)((nTok + 255) / 256)), blk256, 0, stream>>>(enc, eb, nTok);

  // 1. self-attention
  gemm_kernel<1, 2, 0><<<gQKV, blk256, 0, stream>>>(xb, 512, Wq_s, 0, 512L * 64, bq_s, qb,   16384, 512, 512);
  gemm_kernel<1, 2, 0><<<gQKV, blk256, 0, stream>>>(xb, 512, Wk_s, 0, 512L * 64, bk_s, kbuf, 16384, 512, 512);
  gemm_kernel<1, 2, 0><<<gQKV, blk256, 0, stream>>>(xb, 512, Wv_s, 0, 512L * 64, bv_s, vbuf, 16384, 512, 512);
  attn_kernel<<<gAttn, blk128, 0, stream>>>(qb, kbuf, vbuf, mask, ob);
  gemm_kernel<0, 0, 0><<<gProj, blk256, 0, stream>>>(ob, 512, Wo_s, 512, 0, bo_s, t0, 16384, 512, 512);
  ln_kernel<<<dim3(16384), blk256, 0, stream>>>(t0, x, ln1g, ln1b, hf, hb);

  // 2. cross-attention (Q from h, K/V from encoding, no mask)
  gemm_kernel<1, 2, 0><<<gQKV, blk256, 0, stream>>>(hb, 512, Wq_c, 0, 512L * 64, bq_c, qb,   16384, 512, 512);
  gemm_kernel<1, 2, 0><<<gQKV, blk256, 0, stream>>>(eb, 512, Wk_c, 0, 512L * 64, bk_c, kbuf, 16384, 512, 512);
  gemm_kernel<1, 2, 0><<<gQKV, blk256, 0, stream>>>(eb, 512, Wv_c, 0, 512L * 64, bv_c, vbuf, 16384, 512, 512);
  attn_kernel<<<gAttn, blk128, 0, stream>>>(qb, kbuf, vbuf, nullptr, ob);
  gemm_kernel<0, 0, 0><<<gProj, blk256, 0, stream>>>(ob, 512, Wo_c, 512, 0, bo_c, t0, 16384, 512, 512);
  ln_kernel<<<dim3(16384), blk256, 0, stream>>>(t0, hf, ln2g, ln2b, yf, ybuf);

  // 3. FFN + final LN -> d_out (f32)
  gemm_kernel<0, 1, 1><<<gFF1, blk256, 0, stream>>>(ybuf, 512, W1, 2048, 0, b1, fb, 16384, 2048, 512);
  gemm_kernel<0, 0, 0><<<gProj, blk256, 0, stream>>>(fb, 2048, W2, 512, 0, b2, t0, 16384, 512, 2048);
  ln_kernel<<<dim3(16384), blk256, 0, stream>>>(t0, yf, ln3g, ln3b, (float*)d_out, nullptr);

  (void)in_sizes; (void)n_in; (void)out_size; (void)ws_size;
}
